// Transformer_17712445128726
// MI455X (gfx1250) — matrix operand order repaired and checked by reading, not compiled, not run on hardware
//
#include <hip/hip_runtime.h>
#include <hip/hip_bf16.h>
#include <math.h>

// ---------------- model dims (fixed by reference) ----------------
#define B_    2
#define S_    1024
#define D_    512
#define H_    8
#define DFF_  2048
#define DK_   64
#define L_ENC_ 2
#define L_DEC_ 2
#define NEGV  (-1e19f)

typedef __attribute__((ext_vector_type(16))) __bf16 v16bf;
typedef __attribute__((ext_vector_type(8)))  float  v8f;

__device__ __forceinline__ unsigned short f2bf(float f) {
  unsigned int u = __float_as_uint(f);
  u += 0x7FFFu + ((u >> 16) & 1u);          // round-to-nearest-even
  return (unsigned short)(u >> 16);
}

// ---------------- generic batched WMMA GEMM ----------------
// C[z] = act(A[z] @ B[z] + bias) + Res[z];  optional B transpose.
// Requires: K % 64 == 0 (true for all GEMMs in this model: 64/512/1024/2048),
// and all row strides / base offsets even (true for all buffers here).
//
// LDS holds bf16 tiles already permuted into WMMA fragment order
// (ISA 7.12.2 16-bit A 16x32 / B 32x16 layouts), so each operand fetch is a
// single aligned 32B LDS read -> 2x ds_load_b128, no lane repacking.
#define BM 64
#define BN 128
#define BK 64

__global__ __launch_bounds__(256)
void gemm_wmma(const float* __restrict__ A, const float* __restrict__ Bm,
               const float* __restrict__ bias, const float* __restrict__ Res,
               float* __restrict__ C,
               int M, int N, int K, int lda, int ldb, int ldc,
               long long sA, long long sB, long long sC, long long sR,
               int transB, int doGelu)
{
  // [kc][tile][lane][e] fragment-ordered bf16 pairs
  __shared__ __align__(32) unsigned int lA32[BM * BK / 2];   //  8 KB
  __shared__ __align__(32) unsigned int lB32[BK * BN / 2];   // 16 KB

  const int z = blockIdx.z;
  A  += (long long)z * sA;
  Bm += (long long)z * sB;
  C  += (long long)z * sC;
  const float* Rz = Res ? (Res + (long long)z * sR) : (const float*)0;

  const int m0   = blockIdx.y * BM;
  const int n0   = blockIdx.x * BN;
  const int tid  = threadIdx.x;
  const int wid  = tid >> 5;      // 8 waves
  const int lane = tid & 31;
  const int wrow = wid >> 2;      // 0..1 -> 32-row strip
  const int wcol = wid & 3;       // 0..3 -> 32-col strip
  const int half = lane >> 4;
  const int lm   = lane & 15;

  v8f acc[2][2] = {};

  for (int kk = 0; kk < K; kk += BK) {
    // ---- stage A: 64x64 fp32 -> bf16 pairs in fragment order ----
    // pair p = (row r, k-pair kp); consecutive K -> consecutive fragment elems
    #pragma unroll
    for (int it = 0; it < (BM * BK / 2) / 256; ++it) {
      int p  = tid + it * 256;         // 0..2047
      int r  = p >> 5;                 // 0..63
      int k0 = (p & 31) * 2;           // 0..62 even
      float2 v = make_float2(0.0f, 0.0f);
      int gr = m0 + r;
      if (gr < M) v = *(const float2*)(A + (long long)gr * lda + (kk + k0));
      int kc = k0 >> 5, kq = k0 & 31;
      int e0 = (kq & 7) + ((kq >> 4) << 3);                  // A frag element
      int idx = kc * 2048 +
                (((r >> 4) * 32 + ((kq >> 3) & 1) * 16 + (r & 15)) * 16 + e0);
      lA32[idx >> 1] = (unsigned)f2bf(v.x) | ((unsigned)f2bf(v.y) << 16);
    }
    // ---- stage B: 64x128 fp32 -> bf16 pairs in fragment order ----
    #pragma unroll
    for (int it = 0; it < (BK * BN / 2) / 256; ++it) {
      int p  = tid + it * 256;         // 0..4095
      int n  = p & 127;                // col fastest -> coalesced
      int k0 = (p >> 7) * 2;           // 0..62 even
      int gn = n0 + n;
      float2 v = make_float2(0.0f, 0.0f);
      if (gn < N) {
        if (transB) {
          v = *(const float2*)(Bm + (long long)gn * ldb + (kk + k0));
        } else {
          v.x = Bm[(long long)(kk + k0)     * ldb + gn];
          v.y = Bm[(long long)(kk + k0 + 1) * ldb + gn];
        }
      }
      int kc = k0 >> 5, kq = k0 & 31;
      int idx = kc * 4096 +
                (((n >> 4) * 32 + ((kq >> 3) & 1) * 16 + (n & 15)) * 16 + ((kq & 7) + ((kq >> 4) << 3)));
      lB32[idx >> 1] = (unsigned)f2bf(v.x) | ((unsigned)f2bf(v.y) << 16);
    }
    // prefetch next A K-slice (global_prefetch_b8)
    if (kk + BK < K) {
      int gr = m0 + (tid & (BM - 1));
      if (gr < M) __builtin_prefetch((const void*)(A + (long long)gr * lda + kk + BK), 0, 1);
    }
    __syncthreads();

    // ---- compute: operands are contiguous 32B per lane in LDS ----
    const v16bf* pa = (const v16bf*)lA32;   // 128 frags per kc block
    const v16bf* pb = (const v16bf*)lB32;   // 256 frags per kc block
    #pragma unroll
    for (int kc = 0; kc < 2; ++kc) {
      v16bf fa0 = pa[kc * 128 + (wrow * 2 + 0) * 32 + lane];
      v16bf fa1 = pa[kc * 128 + (wrow * 2 + 1) * 32 + lane];
      v16bf fb0 = pb[kc * 256 + (wcol * 2 + 0) * 32 + lane];
      v16bf fb1 = pb[kc * 256 + (wcol * 2 + 1) * 32 + lane];
      acc[0][0] = __builtin_amdgcn_wmma_f32_16x16x32_bf16(false, fa0, false, fb0, (short)0, acc[0][0], false, false);
      acc[0][1] = __builtin_amdgcn_wmma_f32_16x16x32_bf16(false, fa0, false, fb1, (short)0, acc[0][1], false, false);
      acc[1][0] = __builtin_amdgcn_wmma_f32_16x16x32_bf16(false, fa1, false, fb0, (short)0, acc[1][0], false, false);
      acc[1][1] = __builtin_amdgcn_wmma_f32_16x16x32_bf16(false, fa1, false, fb1, (short)0, acc[1][1], false, false);
    }
    __syncthreads();
  }

  // epilogue: bias -> exact GELU -> residual (C layout: VGPR e = row e / e+8)
  #pragma unroll
  for (int ti = 0; ti < 2; ++ti)
    #pragma unroll
    for (int tj = 0; tj < 2; ++tj) {
      int colg = n0 + wcol * 32 + tj * 16 + lm;
      #pragma unroll
      for (int e = 0; e < 8; ++e) {
        int rowg = m0 + wrow * 32 + ti * 16 + e + half * 8;
        if (rowg < M && colg < N) {
          float v = acc[ti][tj][e];
          if (bias)   v += bias[colg];
          if (doGelu) v = 0.5f * v * (1.0f + erff(v * 0.70710678118654752f));
          if (Rz)     v += Rz[(long long)rowg * ldc + colg];
          C[(long long)rowg * ldc + colg] = v;
        }
      }
    }
}

// ---------------- softmax over the QUERY axis (reference quirk) ----------------
// one wave32 per (b,h,k) column; masks + causal + pre/post scaling
__global__ __launch_bounds__(256)
void softmax_q(float* __restrict__ sc, const unsigned char* __restrict__ padq,
               const unsigned char* __restrict__ padk,
               int S, int H, int causal, int scale_after, float inv, int totalCols)
{
  int wid = threadIdx.x >> 5, lane = threadIdx.x & 31;
  int col = blockIdx.x * 8 + wid;
  if (col >= totalCols) return;
  int k  = col % S;
  int bh = col / S;
  int b  = bh / H;
  long long base = (long long)bh * S * S;
  int pk = padk[b * S + k];

  float xv[64];
  int nq = (S + 31) / 32;
  float mx = -3.0e38f;
  for (int i = 0; i < nq; ++i) {
    int q = i * 32 + lane;
    float x = NEGV;
    if (q < S) {
      float v = sc[base + (long long)q * S + k];
      float add = (padq[b * S + q] || pk) ? NEGV : 0.0f;
      if (causal && k > q) add = NEGV;          // min(pad_add, causal_add)
      x = scale_after ? (v + add) : (v + add) * inv;
    }
    xv[i] = x;
    mx = fmaxf(mx, x);
  }
  for (int off = 16; off > 0; off >>= 1) mx = fmaxf(mx, __shfl_xor(mx, off, 32));
  float sum = 0.0f;
  for (int i = 0; i < nq; ++i) { float e = __expf(xv[i] - mx); xv[i] = e; sum += e; }
  for (int off = 16; off > 0; off >>= 1) sum += __shfl_xor(sum, off, 32);
  float scale = (scale_after ? inv : 1.0f) / sum;
  for (int i = 0; i < nq; ++i) {
    int q = i * 32 + lane;
    if (q < S) sc[base + (long long)q * S + k] = xv[i] * scale;
  }
}

// ---------------- elementwise helpers ----------------
// layer norm over the BATCH dim, unbiased std (reference quirk)
__global__ void layernorm_b(const float* __restrict__ x, float* __restrict__ y,
                            int Bn, long long SD)
{
  long long i = (long long)blockIdx.x * 256 + threadIdx.x;
  if (i >= SD) return;
  float mean = 0.0f;
  for (int b = 0; b < Bn; ++b) mean += x[(long long)b * SD + i];
  mean /= (float)Bn;
  float var = 0.0f;
  for (int b = 0; b < Bn; ++b) { float d = x[(long long)b * SD + i] - mean; var += d * d; }
  float sd = sqrtf(var / (float)(Bn - 1));
  float inv = 1.0f / (sd + 1e-4f);
  for (int b = 0; b < Bn; ++b)
    y[(long long)b * SD + i] = (x[(long long)b * SD + i] - mean) * inv;
}

__global__ void posenc(const float* __restrict__ x, float* __restrict__ y,
                       int S, int D, long long total)
{
  long long i = (long long)blockIdx.x * 256 + threadIdx.x;
  if (i >= total) return;
  int d = (int)(i % D);
  long long sd = i / D;
  int s = (int)(sd % S);
  int ii = d >> 1;
  float div = __expf(-((float)(2 * ii) / (float)D) * 9.210340371976184f); // 10000^(-2i/D)
  float ang = (float)s * div;
  float pe = (d & 1) ? cosf(ang) : sinf(ang);
  y[i] = x[i] + pe;
}

__global__ void padmask(const float* __restrict__ x, unsigned char* __restrict__ pad,
                        int n, int D)
{
  int i = blockIdx.x * 256 + threadIdx.x;
  if (i >= n) return;
  const float* p = x + (long long)i * D;
  unsigned char all0 = 1;
  for (int d = 0; d < D; ++d) if (p[d] != 0.0f) { all0 = 0; break; }
  pad[i] = all0;
}

__global__ void build_dec(const float* __restrict__ outp, float* __restrict__ dst,
                          int S, int D, long long total)
{
  long long i = (long long)blockIdx.x * 256 + threadIdx.x;
  if (i >= total) return;
  int d = (int)(i % D);
  long long sd = i / D;
  int s = (int)(sd % S);
  int b = (int)(sd / S);
  dst[i] = (s == 0) ? 1.0f : outp[((long long)b * (S - 1) + (s - 1)) * D + d];
}

__global__ void final_out(const float* __restrict__ cur, const unsigned char* __restrict__ pad,
                          float* __restrict__ o, int S, int D, long long total)
{
  long long i = (long long)blockIdx.x * 256 + threadIdx.x;
  if (i >= total) return;
  int d = (int)(i % D);
  long long sd = i / D;
  int s1 = (int)(sd % (S - 1));
  int b  = (int)(sd / (S - 1));
  int s = s1 + 1;
  o[i] = pad[b * S + s] ? 0.0f : cur[((long long)b * S + s) * D + d];
}

// ---------------- host orchestration ----------------
extern "C" void kernel_launch(void* const* d_in, const int* in_sizes, int n_in,
                              void* d_out, int out_size, void* d_ws, size_t ws_size,
                              hipStream_t stream)
{
  const float* X      = (const float*)d_in[0];
  const float* OutRef = (const float*)d_in[1];
  const float* eComb  = (const float*)d_in[2];
  const float* eHead  = (const float*)d_in[3];
  const float* eWO    = (const float*)d_in[4];
  const float* ebO    = (const float*)d_in[5];
  const float* eW1    = (const float*)d_in[6];
  const float* eW2    = (const float*)d_in[7];
  const float* dsComb = (const float*)d_in[8];
  const float* dsHead = (const float*)d_in[9];
  const float* dsWO   = (const float*)d_in[10];
  const float* dsbO   = (const float*)d_in[11];
  const float* dcComb = (const float*)d_in[12];
  const float* dcHead = (const float*)d_in[13];
  const float* dcWO   = (const float*)d_in[14];
  const float* dcbO   = (const float*)d_in[15];
  const float* dW1    = (const float*)d_in[16];
  const float* dW2    = (const float*)d_in[17];
  float* Yout = (float*)d_out;

  const long long SDt = (long long)B_ * S_ * D_;
  float* w = (float*)d_ws;
  float* cur   = w; w += SDt;
  float* nbuf  = w; w += SDt;
  float* n2buf = w; w += SDt;
  float* rbuf  = w; w += SDt;
  float* encR  = w; w += SDt;
  float* cat   = w; w += SDt;
  float* Qb    = w; w += SDt;     // [B,H,S,DK]
  float* Kb    = w; w += SDt;
  float* Vb    = w; w += SDt;
  float* Tq    = w; w += (long long)B_ * S_ * DK_;
  float* Tk    = w; w += (long long)B_ * S_ * DK_;
  float* Tv    = w; w += (long long)B_ * S_ * DK_;
  float* mid   = w; w += (long long)B_ * S_ * DFF_;
  float* scores = w; w += (long long)B_ * H_ * S_ * S_;
  unsigned char* pad_in  = (unsigned char*)w;
  unsigned char* pad_out = pad_in + B_ * S_;

  auto gemm = [&](const float* A, const float* Bm, const float* bias, const float* Res,
                  float* C, int M, int N, int K, int lda, int ldb, int ldc,
                  long long sA, long long sB, long long sC, long long sR,
                  int transB, int doGelu, int batch) {
    dim3 grid((N + BN - 1) / BN, (M + BM - 1) / BM, batch);
    gemm_wmma<<<grid, 256, 0, stream>>>(A, Bm, bias, Res, C, M, N, K,
                                        lda, ldb, ldc, sA, sB, sC, sR, transB, doGelu);
  };
  auto eg = [](long long n) { return dim3((unsigned)((n + 255) / 256)); };

  auto run_mha = [&](const float* xq, const float* xkv,
                     const float* comb, const float* head,
                     const float* WO, const float* bO,
                     const unsigned char* padq, const unsigned char* padk,
                     int causal, int scale_after,
                     const float* resid, float* rout) {
    // combined projections: [B*S,512]@[512,64]
    gemm(xq,  comb + 0LL * D_ * DK_, 0, 0, Tq, B_ * S_, DK_, D_, D_, DK_, DK_, 0,0,0,0, 0,0, 1);
    gemm(xkv, comb + 1LL * D_ * DK_, 0, 0, Tk, B_ * S_, DK_, D_, D_, DK_, DK_, 0,0,0,0, 0,0, 1);
    gemm(xkv, comb + 2LL * D_ * DK_, 0, 0, Tv, B_ * S_, DK_, D_, D_, DK_, DK_, 0,0,0,0, 0,0, 1);
    // per-head projections: [S,64]@[64,64], batched over b
    for (int h = 0; h < H_; ++h) {
      gemm(Tq, head + (0LL * H_ + h) * DK_ * DK_, 0, 0, Qb + (long long)h * S_ * DK_,
           S_, DK_, DK_, DK_, DK_, DK_,
           (long long)S_ * DK_, 0, (long long)H_ * S_ * DK_, 0, 0,0, B_);
      gemm(Tk, head + (1LL * H_ + h) * DK_ * DK_, 0, 0, Kb + (long long)h * S_ * DK_,
           S_, DK_, DK_, DK_, DK_, DK_,
           (long long)S_ * DK_, 0, (long long)H_ * S_ * DK_, 0, 0,0, B_);
      gemm(Tv, head + (2LL * H_ + h) * DK_ * DK_, 0, 0, Vb + (long long)h * S_ * DK_,
           S_, DK_, DK_, DK_, DK_, DK_,
           (long long)S_ * DK_, 0, (long long)H_ * S_ * DK_, 0, 0,0, B_);
    }
    // scores = Q @ K^T, batched over (b,h)
    gemm(Qb, Kb, 0, 0, scores, S_, S_, DK_, DK_, DK_, S_,
         (long long)S_ * DK_, (long long)S_ * DK_, (long long)S_ * S_, 0, 1, 0, B_ * H_);
    // softmax over query axis (+ masks, scale before/after)
    int totalCols = B_ * H_ * S_;
    softmax_q<<<dim3((totalCols + 7) / 8), 256, 0, stream>>>(
        scores, padq, padk, S_, H_, causal, scale_after, 0.125f, totalCols);
    // attn @ V -> cat[b,q,h,j]
    for (int h = 0; h < H_; ++h) {
      gemm(scores + (long long)h * S_ * S_, Vb + (long long)h * S_ * DK_, 0, 0,
           cat + (long long)h * DK_, S_, DK_, S_, S_, DK_, H_ * DK_,
           (long long)H_ * S_ * S_, (long long)H_ * S_ * DK_, (long long)S_ * H_ * DK_, 0,
           0, 0, B_);
    }
    // output projection + bias + residual (r = a + n)
    gemm(cat, WO, bO, resid, rout, B_ * S_, D_, D_, D_, D_, D_, 0,0,0,0, 0,0, 1);
  };

  // ---------------- encoder ----------------
  padmask<<<eg(B_ * S_), 256, 0, stream>>>(X, pad_in, B_ * S_, D_);
  posenc<<<eg(SDt), 256, 0, stream>>>(X, cur, S_, D_, SDt);
  for (int l = 0; l < L_ENC_; ++l) {
    layernorm_b<<<eg((long long)S_ * D_), 256, 0, stream>>>(cur, nbuf, B_, (long long)S_ * D_);
    run_mha(nbuf, nbuf,
            eComb + (long long)l * 3 * D_ * DK_,
            eHead + (long long)l * 3 * H_ * DK_ * DK_,
            eWO + (long long)l * D_ * D_, ebO + (long long)l * D_,
            pad_in, pad_in, /*causal=*/0, /*scale_after=*/0, nbuf, rbuf);
    layernorm_b<<<eg((long long)S_ * D_), 256, 0, stream>>>(rbuf, n2buf, B_, (long long)S_ * D_);
    gemm(n2buf, eW1 + (long long)l * D_ * DFF_, 0, 0, mid,
         B_ * S_, DFF_, D_, D_, DFF_, DFF_, 0,0,0,0, 0, /*gelu=*/1, 1);
    gemm(mid, eW2 + (long long)l * DFF_ * D_, 0, n2buf, cur,
         B_ * S_, D_, DFF_, DFF_, D_, D_, 0,0,0,0, 0, 0, 1);
  }
  hipMemcpyAsync(encR, cur, SDt * sizeof(float), hipMemcpyDeviceToDevice, stream);

  // ---------------- decoder ----------------
  build_dec<<<eg(SDt), 256, 0, stream>>>(OutRef, rbuf, S_, D_, SDt);
  padmask<<<eg(B_ * S_), 256, 0, stream>>>(rbuf, pad_out, B_ * S_, D_);
  posenc<<<eg(SDt), 256, 0, stream>>>(rbuf, cur, S_, D_, SDt);
  for (int l = 0; l < L_DEC_; ++l) {
    layernorm_b<<<eg((long long)S_ * D_), 256, 0, stream>>>(cur, nbuf, B_, (long long)S_ * D_);
    // self-attn: causal, scale AFTER softmax, pad_in mask (preserved bug)
    run_mha(nbuf, nbuf,
            dsComb + (long long)l * 3 * D_ * DK_,
            dsHead + (long long)l * 3 * H_ * DK_ * DK_,
            dsWO + (long long)l * D_ * D_, dsbO + (long long)l * D_,
            pad_in, pad_in, /*causal=*/1, /*scale_after=*/1, nbuf, rbuf);
    layernorm_b<<<eg((long long)S_ * D_), 256, 0, stream>>>(rbuf, n2buf, B_, (long long)S_ * D_);
    // cross-attn: K/V from encoder repr
    run_mha(n2buf, encR,
            dcComb + (long long)l * 3 * D_ * DK_,
            dcHead + (long long)l * 3 * H_ * DK_ * DK_,
            dcWO + (long long)l * D_ * D_, dcbO + (long long)l * D_,
            pad_out, pad_in, /*causal=*/0, /*scale_after=*/0, n2buf, rbuf);
    layernorm_b<<<eg((long long)S_ * D_), 256, 0, stream>>>(rbuf, nbuf, B_, (long long)S_ * D_);
    gemm(nbuf, dW1 + (long long)l * D_ * DFF_, 0, 0, mid,
         B_ * S_, DFF_, D_, D_, DFF_, DFF_, 0,0,0,0, 0, /*gelu=*/1, 1);
    gemm(mid, dW2 + (long long)l * DFF_ * D_, 0, nbuf, cur,
         B_ * S_, D_, DFF_, DFF_, D_, D_, 0,0,0,0, 0, 0, 1);
  }

  long long totOut = (long long)B_ * (S_ - 1) * D_;
  final_out<<<eg(totOut), 256, 0, stream>>>(cur, pad_out, Yout, S_, D_, totOut);
}
